// MultiHeadAttention_39883066311260
// MI455X (gfx1250) — compile-verified, matrix operand order checked by reading, not run on hardware
//
#include <hip/hip_runtime.h>
#include <hip/hip_bf16.h>
#include <math.h>

// Problem constants
constexpr int NT = 2048;     // tokens
constexpr int DM = 512;      // d_model
constexpr int NH = 8;        // heads
constexpr int DK = 64;       // head dim
constexpr int PK = 5;        // rel buckets
constexpr float NEGV = -9e9f;
constexpr long long NN = (long long)NT * NT;

typedef __attribute__((ext_vector_type(16))) __bf16 v16bf;
typedef __attribute__((ext_vector_type(8)))  __bf16 v8bf;
typedef __attribute__((ext_vector_type(8)))  float  v8f;
typedef __attribute__((ext_vector_type(4)))  unsigned int v4u;
typedef __attribute__((ext_vector_type(8)))  int v8i;
typedef __attribute__((ext_vector_type(4)))  int v4i;

__device__ __forceinline__ v8f wmma16(v16bf a, v16bf b, v8f c) {
  return __builtin_amdgcn_wmma_f32_16x16x32_bf16(false, a, false, b, (short)0, c,
                                                 false, false);
}

// ---------------------------------------------------------------------------
// Fragment loaders. bf16 16x16x32 layout (ISA 7.12.2):
//   A: lane L -> M = L%16, g = L/16; elems e0..7 -> K = g*8+e, e8..15 -> 16+g*8+(e-8)
//   B mirrors with N = L%16. C/D: N = L%16, vgpr r -> M = r + 8*g.
// Caller passes p = rowBase + k0 + g*8; loader reads p[0..7] and p[16..23].
// ---------------------------------------------------------------------------
__device__ __forceinline__ v16bf load_frag_bf16(const __bf16* p) {
  const v8bf a = *(const v8bf*)(p);
  const v8bf b = *(const v8bf*)(p + 16);
  return __builtin_shufflevector(a, b, 0, 1, 2, 3, 4, 5, 6, 7, 8, 9, 10, 11, 12,
                                 13, 14, 15);
}

__device__ __forceinline__ v16bf load_frag_f32(const float* __restrict__ p) {
  const float4 f0 = *(const float4*)(p + 0);
  const float4 f1 = *(const float4*)(p + 4);
  const float4 f2 = *(const float4*)(p + 16);
  const float4 f3 = *(const float4*)(p + 20);
  union { v16bf v; __bf16 e[16]; } u;
  u.e[0]  = (__bf16)f0.x; u.e[1]  = (__bf16)f0.y; u.e[2]  = (__bf16)f0.z; u.e[3]  = (__bf16)f0.w;
  u.e[4]  = (__bf16)f1.x; u.e[5]  = (__bf16)f1.y; u.e[6]  = (__bf16)f1.z; u.e[7]  = (__bf16)f1.w;
  u.e[8]  = (__bf16)f2.x; u.e[9]  = (__bf16)f2.y; u.e[10] = (__bf16)f2.z; u.e[11] = (__bf16)f2.w;
  u.e[12] = (__bf16)f3.x; u.e[13] = (__bf16)f3.y; u.e[14] = (__bf16)f3.z; u.e[15] = (__bf16)f3.w;
  return u.v;
}

// ---------------------------------------------------------------------------
// TDM 2D tile load: 64 rows x 64 bf16 elems -> LDS (row-major, ld=64).
// Descriptor per CDNA5 ISA. This toolchain exposes the
// 6-arg builtin (g0, g1, g2, g3, extra, cpol); groups 2/3 and the extra group
// are zero for a 2D tensor.
// ---------------------------------------------------------------------------
__device__ __forceinline__ void tdm_load_tile64(unsigned int lds_byte_off,
                                                const __bf16* gptr,
                                                unsigned int stride_elems) {
  const unsigned long long ga = (unsigned long long)(uintptr_t)gptr;
  v4u g0;
  g0.x = 1u;                                    // count=1, user descriptor
  g0.y = lds_byte_off;                          // D#.lds_addr (bytes)
  g0.z = (unsigned int)(ga & 0xffffffffu);      // global_addr[31:0]
  g0.w = (unsigned int)((ga >> 32) & 0x1ffffffu) | (2u << 30); // addr[56:32], type=2
  v8i g1;
  g1[0] = (int)(1u << 16);                      // data_size=1 (2 bytes/elem)
  g1[1] = (int)((64u & 0xffffu) << 16);         // tensor_dim0[15:0] @ bits63:48
  g1[2] = (int)((64u >> 16) | ((64u & 0xffffu) << 16)); // dim0 hi / tensor_dim1 lo
  g1[3] = (int)((64u >> 16) | (64u << 16));     // tensor_dim1 hi / tile_dim0=64
  g1[4] = (int)(64u);                           // tile_dim1=64, tile_dim2=0
  g1[5] = (int)stride_elems;                    // tensor_dim0_stride[31:0]
  g1[6] = 0;                                    // stride hi, tensor_dim1_stride lo
  g1[7] = 0;
  const v4i z4 = {0, 0, 0, 0};                  // groups 2/3 unused (2D tensor)
  const v8i z8 = {0, 0, 0, 0, 0, 0, 0, 0};      // extra group (6-arg form)
  __builtin_amdgcn_tensor_load_to_lds(g0, g1, z4, z4, z8, 0);
}

// ---------------------------------------------------------------------------
// TDM double-buffered GEMM: C[M,N] = A(MxK bf16) * B(NxK bf16)^T (+bias).
// Block = 128 thr (4 waves) -> 64x64 C tile; K steps of 64; wave0 drives TDM.
// ---------------------------------------------------------------------------
template <typename TC>
__global__ void gemm_tdm_kernel(const __bf16* __restrict__ A, int lda,
                                const __bf16* __restrict__ B, int ldb,
                                TC* __restrict__ C, int ldc, int Ntiles64, int K,
                                const float* __restrict__ bias) {
  __shared__ __bf16 smem[16384];  // A0 | A1 | B0 | B1, 8KB each
  const int wave = threadIdx.x >> 5;
  const int lane = threadIdx.x & 31;
  const int g = lane >> 4, l16 = lane & 15;
  const int tm64 = blockIdx.x / Ntiles64;
  const int tn64 = blockIdx.x % Ntiles64;
  const int wm = (wave >> 1) * 32, wn = (wave & 1) * 32;
  const int nk = K / 64;
  const __bf16* Ag = A + (size_t)tm64 * 64 * lda;
  const __bf16* Bg = B + (size_t)tn64 * 64 * ldb;

  if (wave == 0) {                       // prologue: fill buffer 0
    tdm_load_tile64(0u, Ag, (unsigned)lda);
    tdm_load_tile64(16384u, Bg, (unsigned)ldb);
  }
  v8f acc[2][2] = {};
  for (int kt = 0; kt < nk; ++kt) {
    const int cur = kt & 1;
    if (wave == 0) {
      if (kt + 1 < nk) {                 // prefetch next K-panel into other buffer
        const int nxt = (kt + 1) & 1;
        tdm_load_tile64((unsigned)(nxt * 8192), Ag + (kt + 1) * 64, (unsigned)lda);
        tdm_load_tile64((unsigned)(16384 + nxt * 8192), Bg + (kt + 1) * 64,
                        (unsigned)ldb);
        __builtin_amdgcn_s_wait_tensorcnt(2);   // current panel's 2 loads done
      } else {
        __builtin_amdgcn_s_wait_tensorcnt(0);
      }
    }
    __syncthreads();                     // panel visible to all waves
    const __bf16* Abuf = smem + cur * 4096;
    const __bf16* Bbuf = smem + 8192 + cur * 4096;
#pragma unroll
    for (int k0 = 0; k0 < 64; k0 += 32) {
      v16bf af[2], bfm[2];
#pragma unroll
      for (int i = 0; i < 2; ++i)
        af[i] = load_frag_bf16(Abuf + (wm + i * 16 + l16) * 64 + k0 + g * 8);
#pragma unroll
      for (int j = 0; j < 2; ++j)
        bfm[j] = load_frag_bf16(Bbuf + (wn + j * 16 + l16) * 64 + k0 + g * 8);
#pragma unroll
      for (int i = 0; i < 2; ++i)
#pragma unroll
        for (int j = 0; j < 2; ++j) acc[i][j] = wmma16(af[i], bfm[j], acc[i][j]);
    }
    __syncthreads();                     // done reading before refill
  }
#pragma unroll
  for (int i = 0; i < 2; ++i)
#pragma unroll
    for (int j = 0; j < 2; ++j) {
      const int col = tn64 * 64 + wn + j * 16 + l16;
      const float bv = bias ? bias[col] : 0.0f;
#pragma unroll
      for (int r = 0; r < 8; ++r) {
        const int row = tm64 * 64 + wm + i * 16 + r + 8 * g;
        C[(size_t)row * ldc + col] = (TC)(acc[i][j][r] + bv);
      }
    }
}

// ---------------------------------------------------------------------------
// Fused score kernel, 16x64 strips: 1 A fragment reused over 4 B tiles.
// S[h,i,j] = mask ? (q_h.k_h + qkrel[h,i,gpm[i,j]])/8 : NEG
// ---------------------------------------------------------------------------
__global__ void score_kernel(const __bf16* __restrict__ q,
                             const __bf16* __restrict__ k,
                             const float* __restrict__ qkr,
                             const int* __restrict__ gpm,
                             const unsigned char* __restrict__ mask,
                             float* __restrict__ S) {
  const int wave = threadIdx.x >> 5;
  const int lane = threadIdx.x & 31;
  const int g = lane >> 4, l16 = lane & 15;
  const int t = blockIdx.x * 4 + wave;   // strip index
  const int G = NT / 64;                 // 32 strip-groups per row band
  const int tm = t / G, tg = t % G;
  const int h = blockIdx.y;

  const __bf16* Ab = q + (size_t)(tm * 16 + l16) * DM + h * DK + g * 8;
  v8f acc[4] = {};
#pragma unroll
  for (int k0 = 0; k0 < DK; k0 += 32) {
    const v16bf a = load_frag_bf16(Ab + k0);
#pragma unroll
    for (int j = 0; j < 4; ++j) {
      const __bf16* Bb =
          k + (size_t)(tg * 64 + j * 16 + l16) * DM + h * DK + g * 8;
      acc[j] = wmma16(a, load_frag_bf16(Bb + k0), acc[j]);
    }
  }
  float* Sh = S + (long long)h * NN;
  const float* qkh = qkr + (long long)h * NT * PK;
#pragma unroll
  for (int j = 0; j < 4; ++j) {
    const int jj = tg * 64 + j * 16 + l16;
#pragma unroll
    for (int r = 0; r < 8; ++r) {
      const int i = tm * 16 + r + 8 * g;
      const long long ij = (long long)i * NT + jj;
      const float bias = qkh[i * PK + gpm[ij]];
      const float val = (acc[j][r] + bias) * 0.125f;  // 1/sqrt(64)
      Sh[ij] = mask[ij] ? val : NEGV;
    }
  }
}

// ---------------------------------------------------------------------------
// AV GEMM, 16x64 strips (full head width): each attn element read exactly once.
// out1[i, h*64+d] = sum_j attn[h,i,j] * vT[h*64+d, j]
// ---------------------------------------------------------------------------
__global__ void av_kernel(const float* __restrict__ S,
                          const __bf16* __restrict__ vT,
                          float* __restrict__ o1) {
  const int wave = threadIdx.x >> 5;
  const int lane = threadIdx.x & 31;
  const int g = lane >> 4, l16 = lane & 15;
  const int tm = blockIdx.x * 4 + wave;  // M tile (16 rows)
  const int h = blockIdx.y;

  const float* Ab = S + (long long)h * NN + (size_t)(tm * 16 + l16) * NT + g * 8;
  v8f acc[4] = {};
  for (int k0 = 0; k0 < NT; k0 += 32) {
    const v16bf a = load_frag_f32(Ab + k0);
#pragma unroll
    for (int j = 0; j < 4; ++j) {
      const __bf16* Bb = vT + (size_t)(h * DK + j * 16 + l16) * NT + g * 8;
      acc[j] = wmma16(a, load_frag_bf16(Bb + k0), acc[j]);
    }
  }
#pragma unroll
  for (int j = 0; j < 4; ++j) {
    const int col = h * DK + j * 16 + l16;
#pragma unroll
    for (int r = 0; r < 8; ++r) {
      const int row = tm * 16 + r + 8 * g;
      o1[(size_t)row * DM + col] = acc[j][r];
    }
  }
}

// ---------------------------------------------------------------------------
// LayerNorm -> bf16. One 256-thread block per row of 512.
// ---------------------------------------------------------------------------
__global__ void ln_kernel(const float* __restrict__ x, const float* __restrict__ w,
                          const float* __restrict__ b, __bf16* __restrict__ y) {
  const int row = blockIdx.x, tid = threadIdx.x;
  const float* xr = x + (size_t)row * DM;
  const float v0 = xr[tid], v1 = xr[tid + 256];
  __shared__ float red[256];
  red[tid] = v0 + v1;
  __syncthreads();
  for (int off = 128; off > 0; off >>= 1) {
    if (tid < off) red[tid] += red[tid + off];
    __syncthreads();
  }
  const float mu = red[0] * (1.0f / DM);
  __syncthreads();
  const float d0 = v0 - mu, d1 = v1 - mu;
  red[tid] = d0 * d0 + d1 * d1;
  __syncthreads();
  for (int off = 128; off > 0; off >>= 1) {
    if (tid < off) red[tid] += red[tid + off];
    __syncthreads();
  }
  const float inv = rsqrtf(red[0] * (1.0f / DM) + 1e-6f);
  __bf16* yr = y + (size_t)row * DM;
  yr[tid]       = (__bf16)(d0 * inv * w[tid] + b[tid]);
  yr[tid + 256] = (__bf16)(d1 * inv * w[tid + 256] + b[tid + 256]);
}

__global__ void cast_bf16_kernel(const float* __restrict__ x,
                                 __bf16* __restrict__ y, int n) {
  const int i = blockIdx.x * blockDim.x + threadIdx.x;
  if (i < n) y[i] = (__bf16)x[i];
}

// qk_rel[h,i,p] = dot(q[i, h*64 : h*64+64], rel_k[p])
__global__ void qkrel_kernel(const __bf16* __restrict__ q,
                             const float* __restrict__ rel_k,
                             float* __restrict__ qkr) {
  const int idx = blockIdx.x * blockDim.x + threadIdx.x;
  if (idx >= NH * NT * PK) return;
  const int p = idx % PK;
  const int i = (idx / PK) % NT;
  const int h = idx / (PK * NT);
  const __bf16* qrow = q + (size_t)i * DM + h * DK;
  const float* rk = rel_k + p * DK;
  float s = 0.0f;
#pragma unroll
  for (int d = 0; d < DK; ++d) s += (float)qrow[d] * rk[d];
  qkr[idx] = s;
}

// v [NT, DM] bf16 -> vT [DM, NT] bf16
__global__ void transpose_kernel(const __bf16* __restrict__ v,
                                 __bf16* __restrict__ vT) {
  const int idx = blockIdx.x * blockDim.x + threadIdx.x; // < NT*DM
  const int i = idx >> 9;
  const int c = idx & 511;
  vT[(size_t)c * NT + i] = v[idx];
}

__device__ __forceinline__ void online_merge(float& m, float& s, float m2, float s2) {
  const float M = fmaxf(m, m2);
  s = s * __expf(m - M) + s2 * __expf(m2 - M);
  m = M;
}

__global__ void rowstats_kernel(const float* __restrict__ S,
                                float* __restrict__ rowm, float* __restrict__ rows) {
  const int bh = blockIdx.x;  // h*NT + i
  const int tid = threadIdx.x;
  const float* Sr = S + (long long)bh * NT;
  float m = -3.0e38f, s = 0.0f;
  for (int j = tid; j < NT; j += 256) {
    const float v = Sr[j];
    const float M = fmaxf(m, v);
    s = s * __expf(m - M) + __expf(v - M);
    m = M;
  }
  __shared__ float sm_[256], ss_[256];
  sm_[tid] = m; ss_[tid] = s;
  __syncthreads();
  for (int off = 128; off > 0; off >>= 1) {
    if (tid < off) {
      float mm = sm_[tid], ssv = ss_[tid];
      online_merge(mm, ssv, sm_[tid + off], ss_[tid + off]);
      sm_[tid] = mm; ss_[tid] = ssv;
    }
    __syncthreads();
  }
  if (tid == 0) { rowm[bh] = sm_[0]; rows[bh] = ss_[0]; }
}

__global__ void colstats_kernel(const float* __restrict__ S,
                                float* __restrict__ colm, float* __restrict__ cols) {
  const int h = blockIdx.x / (NT / 256);
  const int cb = blockIdx.x % (NT / 256);
  const int j = cb * 256 + threadIdx.x;
  const float* Sh = S + (long long)h * NN;
  float m = -3.0e38f, s = 0.0f;
  for (int i = 0; i < NT; ++i) {
    const float v = Sh[(long long)i * NT + j];
    const float M = fmaxf(m, v);
    s = s * __expf(m - M) + __expf(v - M);
    m = M;
  }
  colm[h * NT + j] = m;
  cols[h * NT + j] = s;
}

__global__ void normalize_kernel(float* __restrict__ S,
                                 const float* __restrict__ rowm, const float* __restrict__ rows,
                                 const float* __restrict__ colm, const float* __restrict__ cols) {
  const long long total = (long long)NH * NN;
  const long long stride = (long long)gridDim.x * blockDim.x;
  for (long long idx = (long long)blockIdx.x * blockDim.x + threadIdx.x; idx < total;
       idx += stride) {
    const int h = (int)(idx >> 22);
    const long long rem = idx & (NN - 1);
    const int i = (int)(rem >> 11);
    const int j = (int)(rem & 2047);
    const float v = S[idx];
    const float pr = __expf(v - rowm[h * NT + i]) / rows[h * NT + i];
    const float pc = __expf(v - colm[h * NT + j]) / cols[h * NT + j];
    S[idx] = pr * pc;
  }
}

// gated(bf16) = out1 * sigmoid(gpre)
__global__ void gate_kernel(const float* __restrict__ o1,
                            const float* __restrict__ gpre,
                            __bf16* __restrict__ gated) {
  const int idx = blockIdx.x * blockDim.x + threadIdx.x; // < NT*DM
  const float gp = gpre[idx];
  gated[idx] = (__bf16)(o1[idx] * (1.0f / (1.0f + __expf(-gp))));
}

// ---------------------------------------------------------------------------
extern "C" void kernel_launch(void* const* d_in, const int* in_sizes, int n_in,
                              void* d_out, int out_size, void* d_ws, size_t ws_size,
                              hipStream_t stream) {
  const float* x_q  = (const float*)d_in[0];
  const float* x_k  = (const float*)d_in[1];
  const int*   gpm  = (const int*)d_in[2];
  const unsigned char* mask = (const unsigned char*)d_in[3]; // jax bool -> 1 byte
  const float* ln_w = (const float*)d_in[4];
  const float* ln_b = (const float*)d_in[5];
  const float* Wq   = (const float*)d_in[6];
  const float* Wk   = (const float*)d_in[7];
  const float* Wv   = (const float*)d_in[8];
  const float* Wg   = (const float*)d_in[9];
  const float* bg   = (const float*)d_in[10];
  const float* Wo   = (const float*)d_in[11];
  const float* bo   = (const float*)d_in[12];
  const float* rel_k= (const float*)d_in[13];

  float* out = (float*)d_out;             // [NT, DM]
  float* S   = out + (size_t)NT * DM;     // attn region [NH, NT, NT]

  const size_t ND = (size_t)NT * DM;      // 1,048,576
  const size_t WW = (size_t)DM * DM;      // 262,144
  char* p = (char*)d_ws;
  __bf16* lnq  = (__bf16*)p; p += ND * 2;
  __bf16* lnk  = (__bf16*)p; p += ND * 2;
  __bf16* q    = (__bf16*)p; p += ND * 2;
  __bf16* k    = (__bf16*)p; p += ND * 2;
  __bf16* v    = (__bf16*)p; p += ND * 2;
  __bf16* vT   = (__bf16*)p; p += ND * 2;
  __bf16* gated= (__bf16*)p; p += ND * 2;
  __bf16* Wqb  = (__bf16*)p; p += WW * 2;
  __bf16* Wkb  = (__bf16*)p; p += WW * 2;
  __bf16* Wvb  = (__bf16*)p; p += WW * 2;
  __bf16* Wgb  = (__bf16*)p; p += WW * 2;
  __bf16* Wob  = (__bf16*)p; p += WW * 2;
  float* gpre  = (float*)p;  p += ND * 4;
  float* o1    = (float*)p;  p += ND * 4;
  float* qkr   = (float*)p;  p += (size_t)NH * NT * PK * 4;
  float* rowm  = (float*)p;  p += (size_t)NH * NT * 4;
  float* rows  = (float*)p;  p += (size_t)NH * NT * 4;
  float* colm  = (float*)p;  p += (size_t)NH * NT * 4;
  float* cols  = (float*)p;  p += (size_t)NH * NT * 4;

  // 0) Precast weights to bf16 (read once per GEMM thereafter)
  cast_bf16_kernel<<<(int)(WW / 256), 256, 0, stream>>>(Wq, Wqb, (int)WW);
  cast_bf16_kernel<<<(int)(WW / 256), 256, 0, stream>>>(Wk, Wkb, (int)WW);
  cast_bf16_kernel<<<(int)(WW / 256), 256, 0, stream>>>(Wv, Wvb, (int)WW);
  cast_bf16_kernel<<<(int)(WW / 256), 256, 0, stream>>>(Wg, Wgb, (int)WW);
  cast_bf16_kernel<<<(int)(WW / 256), 256, 0, stream>>>(Wo, Wob, (int)WW);

  // 1) LayerNorm -> bf16
  ln_kernel<<<NT, 256, 0, stream>>>(x_q, ln_w, ln_b, lnq);
  ln_kernel<<<NT, 256, 0, stream>>>(x_k, ln_w, ln_b, lnk);

  // 2) Projections via TDM-tiled GEMM: (2048/64)*(512/64) = 256 blocks
  const int ntl = DM / 64;
  gemm_tdm_kernel<__bf16><<<dim3((NT / 64) * ntl), 128, 0, stream>>>(
      lnq, DM, Wqb, DM, q, DM, ntl, DM, nullptr);
  gemm_tdm_kernel<__bf16><<<dim3((NT / 64) * ntl), 128, 0, stream>>>(
      lnk, DM, Wkb, DM, k, DM, ntl, DM, nullptr);
  gemm_tdm_kernel<__bf16><<<dim3((NT / 64) * ntl), 128, 0, stream>>>(
      lnk, DM, Wvb, DM, v, DM, ntl, DM, nullptr);
  gemm_tdm_kernel<float><<<dim3((NT / 64) * ntl), 128, 0, stream>>>(
      lnq, DM, Wgb, DM, gpre, DM, ntl, DM, bg);

  // 3) V transpose + relative-position projections
  transpose_kernel<<<(int)(ND / 256), 256, 0, stream>>>(v, vT);
  qkrel_kernel<<<(NH * NT * PK + 255) / 256, 256, 0, stream>>>(q, rel_k, qkr);

  // 4) Fused scores -> S (attn region): 4096 strips/head, 4 waves/block
  score_kernel<<<dim3((NT / 16) * (NT / 64) / 4, NH), 128, 0, stream>>>(
      q, k, qkr, gpm, mask, S);

  // 5) Double softmax stats + in-place normalize (S stays in 192MB L2)
  rowstats_kernel<<<NH * NT, 256, 0, stream>>>(S, rowm, rows);
  colstats_kernel<<<NH * (NT / 256), 256, 0, stream>>>(S, colm, cols);
  normalize_kernel<<<32768, 256, 0, stream>>>(S, rowm, rows, colm, cols);

  // 6) out1 = attn @ v per head: 128 strips/head, 4 waves/block
  av_kernel<<<dim3(NT / 16 / 4, NH), 128, 0, stream>>>(S, vT, o1);

  // 7) Gate (bf16) + output projection via TDM GEMM
  gate_kernel<<<(int)(ND / 256), 256, 0, stream>>>(o1, gpre, gated);
  gemm_tdm_kernel<float><<<dim3((NT / 64) * ntl), 128, 0, stream>>>(
      gated, DM, Wob, DM, out, DM, ntl, DM, bo);
}
